// TemplateMatchingMLP_1683627180332
// MI455X (gfx1250) — compile-verified, matrix operand order checked by reading, not run on hardware
//
#include <hip/hip_runtime.h>
#include <hip/hip_bf16.h>

// Problem constants (from reference)
#define VOCAB 8192
#define TSEQ  16
#define HDIM  1024
#define ODIM  8192
#define BATCH 4096
#define TVDIM (TSEQ * VOCAB)   // 131072

typedef __attribute__((ext_vector_type(16))) __bf16 v16bf;
typedef __attribute__((ext_vector_type(8)))  __bf16 bf8;
typedef __attribute__((ext_vector_type(4)))  __bf16 bf4;
typedef __attribute__((ext_vector_type(8)))  float  v8f;

union V16 { v16bf v; bf8 h[2]; };

// ---------------------------------------------------------------------------
// Kernel 1: transpose W1 (H x TV, f32, row-major) -> W1T (TV x H, bf16)
// so the per-token column gather becomes a coalesced row read.
// ---------------------------------------------------------------------------
__global__ __launch_bounds__(256)
void w1_transpose_bf16(const float* __restrict__ W1, __bf16* __restrict__ W1T) {
  __shared__ float tile[32][33];
  const int c0 = blockIdx.x * 32;   // column in TV space
  const int h0 = blockIdx.y * 32;   // row in H space
  for (int r = threadIdx.y; r < 32; r += 8)
    tile[r][threadIdx.x] = W1[(size_t)(h0 + r) * TVDIM + c0 + threadIdx.x];
  __syncthreads();
  for (int r = threadIdx.y; r < 32; r += 8)
    W1T[(size_t)(c0 + r) * HDIM + h0 + threadIdx.x] = (__bf16)tile[threadIdx.x][r];
}

// ---------------------------------------------------------------------------
// Kernel 2: elementwise f32 -> bf16 (for W2, W3, Wout). n4 = n/4.
// ---------------------------------------------------------------------------
__global__ __launch_bounds__(256)
void cvt_f32_bf16(const float* __restrict__ src, __bf16* __restrict__ dst, int n4) {
  int i = blockIdx.x * blockDim.x + threadIdx.x;
  if (i < n4) {
    float4 f = reinterpret_cast<const float4*>(src)[i];
    bf4 o;
    o[0] = (__bf16)f.x; o[1] = (__bf16)f.y; o[2] = (__bf16)f.z; o[3] = (__bf16)f.w;
    reinterpret_cast<bf4*>(dst)[i] = o;
  }
}

// ---------------------------------------------------------------------------
// Kernel 3: embedding gather-sum + bias + relu -> x0 (B x H, bf16).
// One block per batch element; thread t handles 4 consecutive h.
// ---------------------------------------------------------------------------
__global__ __launch_bounds__(256)
void gather_relu(const int* __restrict__ seq, const __bf16* __restrict__ W1T,
                 const float* __restrict__ b1, __bf16* __restrict__ x0) {
  const int b = blockIdx.x;
  __shared__ int sidx[TSEQ];
  if (threadIdx.x < TSEQ) sidx[threadIdx.x] = seq[b * TSEQ + threadIdx.x];
  __syncthreads();
  const int h = threadIdx.x * 4;
  float a0 = b1[h], a1 = b1[h + 1], a2 = b1[h + 2], a3 = b1[h + 3];
#pragma unroll
  for (int t = 0; t < TSEQ; ++t) {
    const __bf16* row = W1T + ((size_t)t * VOCAB + sidx[t]) * HDIM + h;
    bf4 v = *reinterpret_cast<const bf4*>(row);
    a0 += (float)v[0]; a1 += (float)v[1]; a2 += (float)v[2]; a3 += (float)v[3];
  }
  bf4 o;
  o[0] = (__bf16)(a0 > 0.f ? a0 : 0.f);
  o[1] = (__bf16)(a1 > 0.f ? a1 : 0.f);
  o[2] = (__bf16)(a2 > 0.f ? a2 : 0.f);
  o[3] = (__bf16)(a3 > 0.f ? a3 : 0.f);
  *reinterpret_cast<bf4*>(x0 + (size_t)b * HDIM + h) = o;
}

// ---------------------------------------------------------------------------
// Kernel 4: bf16 WMMA GEMM: out = act(A @ W^T + bias)
//   A: M x K bf16 row-major, W: N x K bf16 row-major (so B-matrix col n = W row n)
// Block tile 128x128, K-step 32, 8 waves; wave tile 32x64 = 2x4 WMMA accums.
// LDS pitch 40 bf16 (80 B) -> lane*20 banks mod 64, conflict-free b128 reads.
// ---------------------------------------------------------------------------
template <bool RELU, typename OUT_T>
__global__ __launch_bounds__(256)
void gemm_bf16_wmma(const __bf16* __restrict__ A, const __bf16* __restrict__ W,
                    const float* __restrict__ bias, OUT_T* __restrict__ out,
                    int M, int N, int K) {
  constexpr int BM = 128, BN = 128, BK = 32, PITCH = 40;
  __shared__ __bf16 sA[BM * PITCH];
  __shared__ __bf16 sB[BN * PITCH];

  const int tid  = threadIdx.x;
  const int lane = tid & 31;
  const int wid  = tid >> 5;
  const int wm   = wid & 3;              // 4 wave-strips of 32 rows
  const int wn   = wid >> 2;             // 2 wave-strips of 64 cols
  const int m0   = blockIdx.y * BM;
  const int n0   = blockIdx.x * BN;

  const int lrow = tid >> 1;             // 0..127 (tile row for cooperative load)
  const int lcol = (tid & 1) * 16;       // 0 or 16 (16 bf16 per thread per tile)
  const int sel  = lane >> 4;            // half-wave select (ISA 16-bit K layout)
  const int l15  = lane & 15;

  v8f acc[2][4] = {};

  for (int k0 = 0; k0 < K; k0 += BK) {
    // cooperative global->LDS staging (two b128 loads per thread per tile)
    const uint4* ga = reinterpret_cast<const uint4*>(A + (size_t)(m0 + lrow) * K + k0 + lcol);
    const uint4* gb = reinterpret_cast<const uint4*>(W + (size_t)(n0 + lrow) * K + k0 + lcol);
    uint4 av0 = ga[0], av1 = ga[1];
    uint4 bv0 = gb[0], bv1 = gb[1];
    uint4* pa = reinterpret_cast<uint4*>(sA + lrow * PITCH + lcol);
    uint4* pb = reinterpret_cast<uint4*>(sB + lrow * PITCH + lcol);
    pa[0] = av0; pa[1] = av1;
    pb[0] = bv0; pb[1] = bv1;
    __syncthreads();

    // Fragment assembly per ISA 7.12.2:
    //   lanes 0-15 : K = 0..7 and 16..23 ; lanes 16-31 : K = 8..15 and 24..31
    V16 afrag[2], bfrag[4];
#pragma unroll
    for (int mt = 0; mt < 2; ++mt) {
      const int r = wm * 32 + mt * 16 + l15;
      const bf8* p = reinterpret_cast<const bf8*>(sA + r * PITCH + sel * 8);
      afrag[mt].h[0] = p[0];      // K = sel*8 + 0..7
      afrag[mt].h[1] = p[2];      // K = sel*8 + 16..23
    }
#pragma unroll
    for (int nt = 0; nt < 4; ++nt) {
      const int c = wn * 64 + nt * 16 + l15;
      const bf8* p = reinterpret_cast<const bf8*>(sB + c * PITCH + sel * 8);
      bfrag[nt].h[0] = p[0];
      bfrag[nt].h[1] = p[2];
    }

#pragma unroll
    for (int mt = 0; mt < 2; ++mt)
#pragma unroll
      for (int nt = 0; nt < 4; ++nt)
        acc[mt][nt] = __builtin_amdgcn_wmma_f32_16x16x32_bf16(
            false, afrag[mt].v, false, bfrag[nt].v,
            (short)0, acc[mt][nt], false, false);
    __syncthreads();
  }

  // Epilogue: C layout — VGPR j: lanes 0-15 => M=j, lanes 16-31 => M=j+8, N=lane&15
#pragma unroll
  for (int mt = 0; mt < 2; ++mt) {
#pragma unroll
    for (int nt = 0; nt < 4; ++nt) {
      const int col = n0 + wn * 64 + nt * 16 + l15;
      const float bv = bias[col];
#pragma unroll
      for (int j = 0; j < 8; ++j) {
        const int row = m0 + wm * 32 + mt * 16 + j + sel * 8;
        float v = acc[mt][nt][j] + bv;
        if (RELU) v = v > 0.f ? v : 0.f;
        out[(size_t)row * N + col] = (OUT_T)v;
      }
    }
  }
}

// ---------------------------------------------------------------------------
// Launcher
// ---------------------------------------------------------------------------
extern "C" void kernel_launch(void* const* d_in, const int* in_sizes, int n_in,
                              void* d_out, int out_size, void* d_ws, size_t ws_size,
                              hipStream_t stream) {
  const int*   seq  = (const int*)d_in[0];
  const float* W1   = (const float*)d_in[1];
  const float* b1   = (const float*)d_in[2];
  const float* W2   = (const float*)d_in[3];
  const float* b2   = (const float*)d_in[4];
  const float* W3   = (const float*)d_in[5];
  const float* b3   = (const float*)d_in[6];
  const float* Wout = (const float*)d_in[7];
  const float* bout = (const float*)d_in[8];
  float* out = (float*)d_out;

  // Workspace layout (bytes, all 256B aligned):
  char* ws = (char*)d_ws;
  __bf16* W1T   = (__bf16*)(ws);                                    // 256 MB
  __bf16* W2b   = (__bf16*)(ws + 268435456ull);                     //   2 MB
  __bf16* W3b   = (__bf16*)(ws + 268435456ull + 2097152ull);        //   2 MB
  __bf16* Woutb = (__bf16*)(ws + 268435456ull + 4194304ull);        //  16 MB
  __bf16* X0    = (__bf16*)(ws + 268435456ull + 20971520ull);       //   8 MB
  __bf16* X1    = (__bf16*)(ws + 268435456ull + 29360128ull);       //   8 MB
  __bf16* X2    = (__bf16*)(ws + 268435456ull + 37748736ull);       //   8 MB

  // 1) W1 transpose + bf16 convert
  w1_transpose_bf16<<<dim3(TVDIM / 32, HDIM / 32), dim3(32, 8), 0, stream>>>(W1, W1T);

  // 2) weight conversions
  cvt_f32_bf16<<<(HDIM * HDIM / 4 + 255) / 256, 256, 0, stream>>>(W2, W2b, HDIM * HDIM / 4);
  cvt_f32_bf16<<<(HDIM * HDIM / 4 + 255) / 256, 256, 0, stream>>>(W3, W3b, HDIM * HDIM / 4);
  cvt_f32_bf16<<<(ODIM * HDIM / 4 + 255) / 256, 256, 0, stream>>>(Wout, Woutb, ODIM * HDIM / 4);

  // 3) embedding gather + bias + relu
  gather_relu<<<BATCH, 256, 0, stream>>>(seq, W1T, b1, X0);

  // 4) MLP layers (bf16 WMMA, f32 accumulate)
  gemm_bf16_wmma<true, __bf16><<<dim3(HDIM / 128, BATCH / 128), 256, 0, stream>>>(
      X0, W2b, b2, X1, BATCH, HDIM, HDIM);
  gemm_bf16_wmma<true, __bf16><<<dim3(HDIM / 128, BATCH / 128), 256, 0, stream>>>(
      X1, W3b, b3, X2, BATCH, HDIM, HDIM);
  gemm_bf16_wmma<false, float><<<dim3(ODIM / 128, BATCH / 128), 256, 0, stream>>>(
      X2, Woutb, bout, out, BATCH, ODIM, HDIM);
}
